// VQVAE_34_57019985822237
// MI455X (gfx1250) — matrix operand order repaired and checked by reading, not compiled, not run on hardware
//
#include <hip/hip_runtime.h>
#include <hip/hip_bf16.h>
#include <math.h>

typedef __attribute__((ext_vector_type(16))) __bf16 bf16x16;
typedef __attribute__((ext_vector_type(8)))  __bf16 bf16x8;
typedef __attribute__((ext_vector_type(4)))  __bf16 bf16x4;
typedef __attribute__((ext_vector_type(8)))  float  f32x8;

#define BM 128
#define BN 64
#define BK 64

// ---------------------------------------------------------------------------
// weight transpose: w[OC][IC][Kw] f32  ->  wt[(k*IC+i)][OC] bf16
// ---------------------------------------------------------------------------
__global__ __launch_bounds__(256)
void k_wtrans(const float* __restrict__ w, __bf16* __restrict__ wt,
              int OC, int IC, int Kw)
{
    long total  = (long)OC * IC * Kw;
    long stride = (long)gridDim.x * blockDim.x;
    for (long i = (long)blockIdx.x * blockDim.x + threadIdx.x; i < total; i += stride) {
        int  k = (int)(i % Kw);
        long r = i / Kw;
        int  c = (int)(r % IC);
        int  o = (int)(r / IC);
        wt[((long)k * IC + c) * OC + o] = (__bf16)w[i];
    }
}

// ---------------------------------------------------------------------------
// Implicit-GEMM conv1d (channels-last), bf16 WMMA, f32 accumulate.
//   A row m=(n,t): concat over taps k of X[n][t*stride+k*dil-pad][0..IC)
//   B = Wt[Kw*IC][OC];  Y[n][t][oc] = GEMM + bias (+resid) (+relu)
// ups2: nearest x2 upsample folded into the A-gather.
// icl2:  log2(IC)  if IC is pow2 and IC%BK==0, else -1 (generic path)
// toutl2: log2(Tout) (Tout is always pow2 in this net)
// ---------------------------------------------------------------------------
__global__ __launch_bounds__(256)
void k_conv_wmma(const float* __restrict__ X, const __bf16* __restrict__ Wt,
                 const float* __restrict__ bias, const float* __restrict__ resid,
                 float* __restrict__ Y,
                 int Nb, int Tin, int Tout, int IC, int OC,
                 int Kw, int stride, int dil, int pad,
                 int ups2, int relu_in, int relu_out,
                 int icl2, int toutl2)
{
    alignas(16) __shared__ __bf16 As[BM][BK + 8];   // row stride 144B (16B mult)
    alignas(16) __shared__ __bf16 Bs[BN][BK + 8];   // [col][k] transposed store

    const int Ktot  = Kw * IC;
    const int M     = Nb * Tout;
    const int mtile = blockIdx.x * BM;
    const int ntile = blockIdx.y * BN;

    const int tid  = threadIdx.x;
    const int lane = tid & 31;
    const int wave = tid >> 5;
    const int wm   = wave & 3;      // 4 waves along M
    const int wn   = wave >> 2;     // 2 waves along N

    const int lrow = lane & 15;
    const int hi   = lane >> 4;     // 0 | 1
    const int kbA  = hi ? 8  : 0;   // A lane K-base (16-bit A 16x32 layout)
    const int kbB  = hi ? 8  : 0;   // B lane K-base (matches A K order)
    const int TinEff = ups2 ? (Tin << 1) : Tin;

    const bool aFast = (icl2 >= 0);                                   // IC pow2, IC%BK==0
    const bool bFast = ((OC & (BN - 1)) == 0) && ((Ktot & (BK - 1)) == 0);

    // ---- K-invariant per-thread A-row descriptors (fast path: float4 cols) ----
    const int  c4     = (tid & 15) * 4;          // channel-quad column in tile
    const float* abase[8];
    int        atb[8];
    if (aFast) {
        #pragma unroll
        for (int i = 0; i < 8; ++i) {
            int r = (tid >> 4) + 16 * i;         // tile rows 0..127
            int m = mtile + r;
            int n = m >> toutl2;
            int t = m & (Tout - 1);
            abase[i] = X + (long)n * Tin * IC;
            atb[i]   = t * stride - pad;
        }
    }

    const f32x8 z8 = {0.f,0.f,0.f,0.f,0.f,0.f,0.f,0.f};
    f32x8 acc[2][2];
    acc[0][0]=z8; acc[0][1]=z8; acc[1][0]=z8; acc[1][1]=z8;

    for (int k0 = 0; k0 < Ktot; k0 += BK) {
        // ================= stage A tile (BM x BK) =================
        if (aFast) {
            int kg  = k0 + c4;
            int tap = kg >> icl2;
            int ch  = kg & (IC - 1);
            #pragma unroll
            for (int i = 0; i < 8; ++i) {
                int r  = (tid >> 4) + 16 * i;
                int ti = atb[i] + tap * dil;
                float4 v = make_float4(0.f, 0.f, 0.f, 0.f);
                if (ti >= 0 && ti < TinEff) {
                    int tis = ups2 ? (ti >> 1) : ti;
                    v = *(const float4*)(abase[i] + (long)tis * IC + ch);
                    if (relu_in) {
                        v.x = fmaxf(v.x, 0.f); v.y = fmaxf(v.y, 0.f);
                        v.z = fmaxf(v.z, 0.f); v.w = fmaxf(v.w, 0.f);
                    }
                }
                bf16x4 b; b[0]=(__bf16)v.x; b[1]=(__bf16)v.y; b[2]=(__bf16)v.z; b[3]=(__bf16)v.w;
                *(bf16x4*)&As[r][c4] = b;
            }
        } else {
            for (int e = tid; e < BM*BK; e += 256) {
                int r = e >> 6, c = e & 63;
                int kg = k0 + c;
                float v = 0.f;
                int m = mtile + r;
                if (m < M && kg < Ktot) {
                    int n   = m >> toutl2;
                    int t   = m & (Tout - 1);
                    int tap = kg / IC;
                    int ch  = kg - tap * IC;
                    int ti  = t*stride + tap*dil - pad;
                    if (ti >= 0 && ti < TinEff) {
                        if (ups2) ti >>= 1;
                        v = X[((long)n*Tin + ti)*IC + ch];
                        if (relu_in) v = fmaxf(v, 0.f);
                    }
                }
                As[r][c] = (__bf16)v;
            }
        }
        // ================= stage B tile (BK x BN) -> Bs[col][k] =================
        if (bFast) {
            #pragma unroll
            for (int i = 0; i < 4; ++i) {
                int k = (tid >> 4) + 16 * i;     // k rows 0..63
                const __bf16* src = &Wt[(long)(k0 + k) * OC + ntile + c4];
                bf16x4 w = *(const bf16x4*)src;
                Bs[c4 + 0][k] = w[0];
                Bs[c4 + 1][k] = w[1];
                Bs[c4 + 2][k] = w[2];
                Bs[c4 + 3][k] = w[3];
            }
            if (k0 + BK < Ktot)   // hint next weight tile toward L2/L0
                __builtin_prefetch(&Wt[(long)(k0 + BK + (tid >> 4)) * OC + ntile + c4], 0, 1);
        } else {
            for (int e = tid; e < BK*BN; e += 256) {
                int c = e & 63, r = e >> 6;
                int kg = k0 + r, col = ntile + c;
                __bf16 v = (__bf16)0.f;
                if (kg < Ktot && col < OC) v = Wt[(long)kg*OC + col];
                Bs[c][r] = v;
            }
        }
        __syncthreads();

        // ================= 2 x (2x2) WMMA over the BK=64 slab =================
        #pragma unroll
        for (int kk = 0; kk < BK; kk += 32) {
            union { bf16x16 v; bf16x8 h[2]; } af[2], bfg[2];
            #pragma unroll
            for (int mi = 0; mi < 2; ++mi) {
                int rr = wm*32 + mi*16 + lrow;
                af[mi].h[0] = *(const bf16x8*)&As[rr][kk + kbA];
                af[mi].h[1] = *(const bf16x8*)&As[rr][kk + kbA + 16];
            }
            #pragma unroll
            for (int ni = 0; ni < 2; ++ni) {
                int cc = wn*32 + ni*16 + lrow;
                bfg[ni].h[0] = *(const bf16x8*)&Bs[cc][kk + kbB];
                bfg[ni].h[1] = *(const bf16x8*)&Bs[cc][kk + kbB + 16];
            }
            #pragma unroll
            for (int mi = 0; mi < 2; ++mi)
                #pragma unroll
                for (int ni = 0; ni < 2; ++ni)
                    acc[mi][ni] = __builtin_amdgcn_wmma_f32_16x16x32_bf16(
                        false, af[mi].v, false, bfg[ni].v,
                        (short)0, acc[mi][ni], false, false);
        }
        __syncthreads();
    }

    // ---- epilogue: D layout = elem r in lane L -> row r+8*(L>=16), col L&15
    const int rofs = hi * 8;
    #pragma unroll
    for (int mi = 0; mi < 2; ++mi)
        #pragma unroll
        for (int ni = 0; ni < 2; ++ni) {
            int col = ntile + wn*32 + ni*16 + lrow;
            #pragma unroll
            for (int r = 0; r < 8; ++r) {
                int row = mtile + wm*32 + mi*16 + rofs + r;
                if (row < M && col < OC) {
                    float v = acc[mi][ni][r];
                    if (bias)     v += bias[col];
                    if (resid)    v += resid[(long)row*OC + col];
                    if (relu_out) v = fmaxf(v, 0.f);
                    Y[(long)row*OC + col] = v;
                }
            }
        }
}

// ---------------------------------------------------------------------------
// VQ helper kernels
// ---------------------------------------------------------------------------
__global__ __launch_bounds__(256)
void k_zero_i32(int* __restrict__ p, int n)
{
    int i = blockIdx.x * blockDim.x + threadIdx.x;
    if (i < n) p[i] = 0;
}

__global__ __launch_bounds__(256)
void k_cnorm(const float* __restrict__ cb, float* __restrict__ cnorm, int K, int C)
{
    int j = blockIdx.x * blockDim.x + threadIdx.x;
    if (j < K) {
        float s = 0.f;
        for (int c = 0; c < C; ++c) { float v = cb[(long)j*C + c]; s += v*v; }
        cnorm[j] = s;
    }
}

// per-row (one block per m): argmin_j (||c_j||^2 - 2*S[m][j]);
// gather Q[m] = codebook[idx]; accumulate commit loss and code histogram.
__global__ __launch_bounds__(256)
void k_argmin_quant(const float* __restrict__ S, const float* __restrict__ cnorm,
                    const float* __restrict__ cb, const float* __restrict__ E,
                    float* __restrict__ Q, int* __restrict__ idx_out,
                    int* __restrict__ hist, float* __restrict__ commit,
                    int C, int K)
{
    __shared__ float sval[256];
    __shared__ int   sidx[256];
    const int m = blockIdx.x, tid = threadIdx.x;

    float best = 3.4e38f; int bi = 0;
    for (int j = tid; j < K; j += 256) {
        float d = cnorm[j] - 2.f * S[(long)m*K + j];
        if (d < best) { best = d; bi = j; }
    }
    sval[tid] = best; sidx[tid] = bi;
    __syncthreads();
    for (int s = 128; s > 0; s >>= 1) {
        if (tid < s) {
            float ov = sval[tid+s]; int oi = sidx[tid+s];
            if (ov < sval[tid] || (ov == sval[tid] && oi < sidx[tid])) {
                sval[tid] = ov; sidx[tid] = oi;
            }
        }
        __syncthreads();
    }
    const int j = sidx[0];
    if (tid == 0) { idx_out[m] = j; atomicAdd(&hist[j], 1); }

    float csum = 0.f;
    for (int c = tid; c < C; c += 256) {
        float q = cb[(long)j*C + c];
        Q[(long)m*C + c] = q;
        float d = E[(long)m*C + c] - q;
        csum += d*d;
    }
    __syncthreads();
    sval[tid] = csum;
    __syncthreads();
    for (int s = 128; s > 0; s >>= 1) {
        if (tid < s) sval[tid] += sval[tid+s];
        __syncthreads();
    }
    if (tid == 0) atomicAdd(commit, sval[0]);
}

__global__ __launch_bounds__(256)
void k_finalize(const int* __restrict__ hist, const float* __restrict__ commit,
                float* __restrict__ out2, int K, int NT, int MC)
{
    __shared__ float s[256];
    int tid = threadIdx.x;
    float acc = 0.f;
    for (int j = tid; j < K; j += 256) {
        float p = (float)hist[j] / (float)NT;
        acc += p * logf(p + 1e-7f);
    }
    s[tid] = acc;
    __syncthreads();
    for (int st = 128; st > 0; st >>= 1) {
        if (tid < st) s[tid] += s[tid+st];
        __syncthreads();
    }
    if (tid == 0) {
        out2[0] = *commit / (float)MC;   // commit loss (mean)
        out2[1] = expf(-s[0]);           // perplexity
    }
}

// ---------------------------------------------------------------------------
// host orchestration
// ---------------------------------------------------------------------------
static inline int ilog2_exact(int v)   // log2 if pow2, else -1
{
    int s = 0;
    while ((1 << s) < v) ++s;
    return ((1 << s) == v) ? s : -1;
}

static inline void run_conv(hipStream_t s, const float* X, const float* w, const float* b,
                            const float* resid, float* Y, __bf16* wt,
                            int Nb, int Tin, int Tout, int IC, int OC,
                            int Kw, int stride, int dil, int pad,
                            int ups2, int relu_in, int relu_out)
{
    long wn = (long)OC * IC * Kw;
    int  tb = (int)((wn + 255) / 256); if (tb > 2048) tb = 2048;
    k_wtrans<<<tb, 256, 0, s>>>(w, wt, OC, IC, Kw);
    int M = Nb * Tout;
    int icl2 = ilog2_exact(IC);
    if (icl2 >= 0 && (IC % BK) != 0) icl2 = -1;   // fast path needs IC%BK==0
    int toutl2 = ilog2_exact(Tout);
    dim3 grid((M + BM - 1) / BM, (OC + BN - 1) / BN);
    k_conv_wmma<<<grid, 256, 0, s>>>(X, wt, b, resid, Y, Nb, Tin, Tout, IC, OC,
                                     Kw, stride, dil, pad, ups2, relu_in, relu_out,
                                     icl2, toutl2);
}

extern "C" void kernel_launch(void* const* d_in, const int* in_sizes, int n_in,
                              void* d_out, int out_size, void* d_ws, size_t ws_size,
                              hipStream_t stream)
{
    (void)in_sizes; (void)n_in;
    const int Nb = 16, IN_DIM = 34, W = 512, CODE = 512, NCODE = 1024;

    // ---- unpack inputs (setup_inputs() dict insertion order, recursive) ----
    int ii = 0;
    auto nf = [&]() { return (const float*)d_in[ii++]; };
    struct ResP  { const float *w1,*b1,*w2,*b2; };
    struct DownP { const float *dw,*db; ResP res[3]; };
    struct UpP   { ResP res[3]; const float *uw,*ub; };

    const float* x = nf();
    const float *e_in_w = nf(), *e_in_b = nf(), *e_out_w = nf(), *e_out_b = nf();
    DownP downs[3];
    for (int b = 0; b < 3; ++b) {
        downs[b].dw = nf(); downs[b].db = nf();
        for (int r = 0; r < 3; ++r) {
            downs[b].res[r].w1 = nf(); downs[b].res[r].b1 = nf();
            downs[b].res[r].w2 = nf(); downs[b].res[r].b2 = nf();
        }
    }
    const float *dd_in_w = nf(), *dd_in_b = nf();
    const float *dd_mid_w = nf(), *dd_mid_b = nf();
    const float *dd_out_w = nf(), *dd_out_b = nf();
    UpP ups[3];
    for (int b = 0; b < 3; ++b) {
        for (int r = 0; r < 3; ++r) {
            ups[b].res[r].w1 = nf(); ups[b].res[r].b1 = nf();
            ups[b].res[r].w2 = nf(); ups[b].res[r].b2 = nf();
        }
        ups[b].uw = nf(); ups[b].ub = nf();
    }
    const float* codebook = nf();

    // ---- workspace layout ----
    const size_t ACT = (size_t)Nb * 512 * W * sizeof(float);        // 16 MiB
    const size_t WTB = (size_t)4 << 20;                             // 4 MiB
    const size_t SCB = (size_t)NCODE * (Nb * 64) * sizeof(float);   // 4 MiB
    char* ws = (char*)d_ws;
    float*  bufs[3] = { (float*)ws, (float*)(ws + ACT), (float*)(ws + 2*ACT) };
    __bf16* wt      = (__bf16*)(ws + 3*ACT);
    float*  scores  = (float*)(ws + 3*ACT + WTB);
    int*    idxbuf  = (int*)(ws + 3*ACT + WTB + SCB);
    int*    hist    = idxbuf + 1024;
    float*  commit  = (float*)(hist + 1024);        // contiguous after hist
    float*  cnorm   = commit + 16;
    const size_t NEED = 3*ACT + WTB + SCB + (2048 + 16 + NCODE) * sizeof(float);
    if (ws_size < NEED) return;

    const int DIL[3] = {9, 3, 1};

    // ---------------- encoder ----------------
    run_conv(stream, x, e_in_w, e_in_b, nullptr, bufs[0], wt,
             Nb, 512, 512, IN_DIM, W, 3,1,1,1, 0, 0, 1);     // h = relu(conv_in(x))
    int cur = 0, T = 512;
    for (int b = 0; b < 3; ++b) {
        int Tout = T / 2;
        int dst = (cur + 1) % 3;
        run_conv(stream, bufs[cur], downs[b].dw, downs[b].db, nullptr, bufs[dst], wt,
                 Nb, T, Tout, W, W, 4,2,1,1, 0, 0, 0);
        cur = dst; T = Tout;
        for (int r = 0; r < 3; ++r) {
            int d = DIL[r];
            int t1 = (cur + 1) % 3, t2 = (cur + 2) % 3;
            run_conv(stream, bufs[cur], downs[b].res[r].w1, downs[b].res[r].b1, nullptr,
                     bufs[t1], wt, Nb, T, T, W, W, 3,1,d,d, 0, 1, 0);
            run_conv(stream, bufs[t1], downs[b].res[r].w2, downs[b].res[r].b2, bufs[cur],
                     bufs[t2], wt, Nb, T, T, W, W, 1,1,1,0, 0, 1, 0);
            cur = t2;
        }
    }
    int eidx = (cur + 1) % 3;
    run_conv(stream, bufs[cur], e_out_w, e_out_b, nullptr, bufs[eidx], wt,
             Nb, T, T, W, CODE, 3,1,1,1, 0, 0, 0);
    float* E = bufs[eidx];                                  // (16,64,512)

    // ---------------- VQ ----------------
    const int NT = Nb * T;                                  // 1024 rows
    run_conv(stream, E, codebook, nullptr, nullptr, scores, wt,
             Nb, T, T, CODE, NCODE, 1,1,1,0, 0, 0, 0);      // E @ codebook^T
    k_cnorm<<<(NCODE + 255) / 256, 256, 0, stream>>>(codebook, cnorm, NCODE, CODE);
    k_zero_i32<<<(1025 + 255) / 256, 256, 0, stream>>>(hist, 1025);   // hist + commit
    int qidx = (eidx + 1) % 3;
    float* Q = bufs[qidx];
    k_argmin_quant<<<NT, 256, 0, stream>>>(scores, cnorm, codebook, E, Q,
                                           idxbuf, hist, commit, CODE, NCODE);
    cur = qidx;

    // ---------------- decoder ----------------
    int dst = (cur + 1) % 3;
    run_conv(stream, bufs[cur], dd_in_w, dd_in_b, nullptr, bufs[dst], wt,
             Nb, T, T, CODE, W, 3,1,1,1, 0, 0, 1);          // h = relu(conv_in(xq))
    cur = dst;
    for (int b = 0; b < 3; ++b) {
        for (int r = 0; r < 3; ++r) {
            int d = DIL[r];
            int t1 = (cur + 1) % 3, t2 = (cur + 2) % 3;
            run_conv(stream, bufs[cur], ups[b].res[r].w1, ups[b].res[r].b1, nullptr,
                     bufs[t1], wt, Nb, T, T, W, W, 3,1,d,d, 0, 1, 0);
            run_conv(stream, bufs[t1], ups[b].res[r].w2, ups[b].res[r].b2, bufs[cur],
                     bufs[t2], wt, Nb, T, T, W, W, 1,1,1,0, 0, 1, 0);
            cur = t2;
        }
        int Tout = T * 2;
        int d2 = (cur + 1) % 3;
        run_conv(stream, bufs[cur], ups[b].uw, ups[b].ub, nullptr, bufs[d2], wt,
                 Nb, T, Tout, W, W, 3,1,1,1, 1, 0, 0);      // x2 upsample folded in
        cur = d2; T = Tout;
    }
    int d3 = (cur + 1) % 3;
    run_conv(stream, bufs[cur], dd_mid_w, dd_mid_b, nullptr, bufs[d3], wt,
             Nb, T, T, W, W, 3,1,1,1, 0, 0, 1);             // h = relu(conv_mid(h))
    cur = d3;
    float* out = (float*)d_out;
    run_conv(stream, bufs[cur], dd_out_w, dd_out_b, nullptr, out, wt,
             Nb, T, T, W, IN_DIM, 3,1,1,1, 0, 0, 0);        // (N,T,34) == x_out

    k_finalize<<<1, 256, 0, stream>>>(hist, commit, out + (out_size - 2),
                                      NCODE, NT, NT * CODE);
}
